// GNN_23338852287245
// MI455X (gfx1250) — compile-verified, matrix operand order checked by reading, not run on hardware
//
#include <hip/hip_runtime.h>
#include <math.h>

typedef __attribute__((ext_vector_type(4)))  _Float16 v4h;
typedef __attribute__((ext_vector_type(8)))  _Float16 v8h;
typedef __attribute__((ext_vector_type(16))) _Float16 v16h;
typedef __attribute__((ext_vector_type(8)))  float    v8f;

#define DIN 64
#define LXS 72   // LDS row stride in halves: 144B = 16B-aligned, 36 dwords -> no bank conflicts

static __device__ __forceinline__ v16h cat8(v8h a, v8h b) {
    return __builtin_shufflevector(a, b, 0,1,2,3,4,5,6,7,8,9,10,11,12,13,14,15);
}

// ISA 7.12.2 16-bit fragment: VGPRs 0-3 hold K = kb + hi*8 .. +7 (8 contiguous
// halves = 16B), VGPRs 4-7 hold K = kb+16 + hi*8 .. +7  -> two ds_load_b128.
static __device__ __forceinline__ v16h load_frag(const _Float16* row, int kb, int hi) {
    const v8h g0 = *(const v8h*)(row + kb + hi * 8);
    const v8h g1 = *(const v8h*)(row + kb + 16 + hi * 8);
    return cat8(g0, g1);
}

// ---------------------------------------------------------------------------
// Fused q/k/v/skip projection GEMM: O_p = X @ W_p + b_p  for p in 0..3.
// Block = 256 threads = 8 waves; block covers (2048/DOUT) rows x DOUT cols.
// A fragments built once and reused across all four projections.
// ---------------------------------------------------------------------------
template <int DOUT>
__global__ __launch_bounds__(256) void qkvs_gemm(
    const float* __restrict__ X,
    const float* __restrict__ W0, const float* __restrict__ W1,
    const float* __restrict__ W2, const float* __restrict__ W3,
    const float* __restrict__ B0, const float* __restrict__ B1,
    const float* __restrict__ B2, const float* __restrict__ B3,
    float* __restrict__ O0, float* __restrict__ O1,
    float* __restrict__ O2, float* __restrict__ O3)
{
    constexpr int NT   = DOUT / 16;   // col tiles: 4 (DOUT=64) or 2 (DOUT=32)
    constexpr int RT   = 8 / NT;      // row tiles: 2 or 4
    constexpr int ROWS = RT * 16;     // rows per block: 32 or 64

    __shared__ alignas(16) _Float16 lx [ROWS * LXS];  // X tile, f16
    __shared__ alignas(16) _Float16 lwT[DOUT * LXS];  // W^T tile, f16 (n-major)

    const int tid  = threadIdx.x;
    const int lane = tid & 31;
    const int wv   = tid >> 5;
    const int rt   = wv / NT, ct = wv % NT;
    const int rowl = lane & 15;
    const int hi   = lane >> 4;
    const int m0g  = blockIdx.x * ROWS;

    // ---- load X tile as float4, convert to f16, b64 LDS stores -------------
    for (int i = tid; i < ROWS * (DIN / 4); i += 256) {
        int r = i >> 4, c = i & 15;
        float4 xv = ((const float4*)X)[(size_t)(m0g + r) * (DIN / 4) + c];
        v4h hv = { (_Float16)xv.x, (_Float16)xv.y, (_Float16)xv.z, (_Float16)xv.w };
        *(v4h*)&lx[r * LXS + c * 4] = hv;
    }
    __syncthreads();

    // ---- A fragments: built once, live in registers for all projections ---
    const _Float16* arow = &lx[(rt * 16 + rowl) * LXS];
    const v16h a0 = load_frag(arow, 0,  hi);
    const v16h a1 = load_frag(arow, 32, hi);

    const float* Ws[4] = {W0, W1, W2, W3};
    const float* Bs[4] = {B0, B1, B2, B3};
    float*       Os[4] = {O0, O1, O2, O3};

    #pragma unroll
    for (int p = 0; p < 4; ++p) {
        // ---- load W (row-major [64][DOUT]) transposed into LDS ------------
        for (int i = tid; i < (DIN * DOUT) / 4; i += 256) {
            float4 wv = ((const float4*)Ws[p])[i];
            int r = (i * 4) / DOUT;          // k
            int c = (i * 4) % DOUT;          // n
            lwT[(c + 0) * LXS + r] = (_Float16)wv.x;
            lwT[(c + 1) * LXS + r] = (_Float16)wv.y;
            lwT[(c + 2) * LXS + r] = (_Float16)wv.z;
            lwT[(c + 3) * LXS + r] = (_Float16)wv.w;
        }
        __syncthreads();

        // ---- B fragments: contiguous b128 loads from transposed tile ------
        const _Float16* brow = &lwT[(ct * 16 + rowl) * LXS];
        const v16h b0 = load_frag(brow, 0,  hi);
        const v16h b1 = load_frag(brow, 32, hi);

        v8f acc = {};
        acc = __builtin_amdgcn_wmma_f32_16x16x32_f16(false, a0, false, b0,
                                                     (short)0, acc, false, false);
        acc = __builtin_amdgcn_wmma_f32_16x16x32_f16(false, a1, false, b1,
                                                     (short)0, acc, false, false);

        // ---- bias + store: immediate-offset global stores -----------------
        // C/D layout: VGPR r -> M = r + hi*8, N = lane&15
        int   n  = ct * 16 + rowl;
        float bb = Bs[p][n];
        float* op = Os[p] + (size_t)(m0g + rt * 16 + hi * 8) * DOUT + n;
        #pragma unroll
        for (int r8 = 0; r8 < 8; ++r8)
            op[r8 * DOUT] = acc[r8] + bb;
        __syncthreads();   // before next projection overwrites lwT
    }
}

// ---------------------------------------------------------------------------
// Edge stage: segment softmax + weighted aggregation via L2-resident atomics.
// ---------------------------------------------------------------------------
__global__ void zero_f32(float* __restrict__ p, long n) {
    long i = (long)blockIdx.x * blockDim.x + threadIdx.x;
    if (i < n) p[i] = 0.0f;
}

// order-preserving uint encoding of float for atomic max
__device__ __forceinline__ unsigned flipf(float f) {
    unsigned u = __float_as_uint(f);
    return (u & 0x80000000u) ? ~u : (u | 0x80000000u);
}
__device__ __forceinline__ float unflipf(unsigned u) {
    return __uint_as_float((u & 0x80000000u) ? (u & 0x7fffffffu) : ~u);
}

// one wave per edge: logits[e] = dot(q[dst], k[src]) * scale ; atomicMax m[dst]
template <int D>
__global__ __launch_bounds__(256) void edge_logits(
    const float* __restrict__ q, const float* __restrict__ k,
    const int* __restrict__ src, const int* __restrict__ dst,
    float* __restrict__ logits, unsigned* __restrict__ mflip,
    int E, float scale)
{
    int e    = blockIdx.x * 8 + (threadIdx.x >> 5);
    int lane = threadIdx.x & 31;
    if (e >= E) return;                 // uniform per wave
    int s = src[e], t = dst[e];
    float acc;
    if constexpr (D == 64) {
        float2 qv = ((const float2*)(q + (size_t)t * 64))[lane];
        float2 kv = ((const float2*)(k + (size_t)s * 64))[lane];
        acc = qv.x * kv.x + qv.y * kv.y;
    } else {
        acc = q[(size_t)t * D + lane] * k[(size_t)s * D + lane];
    }
    #pragma unroll
    for (int off = 16; off > 0; off >>= 1)
        acc += __shfl_xor(acc, off);
    if (lane == 0) {
        float lg = acc * scale;
        logits[e] = lg;
        atomicMax(&mflip[t], flipf(lg));
    }
}

// thread per edge: ex[e] = exp(logit - m[dst]); denom[dst] += ex
__global__ void edge_exp(
    const float* __restrict__ logits, const unsigned* __restrict__ mflip,
    const int* __restrict__ dst, float* __restrict__ exw,
    float* __restrict__ denom, int E)
{
    int e = blockIdx.x * blockDim.x + threadIdx.x;
    if (e >= E) return;
    int t = dst[e];
    float ex = __expf(logits[e] - unflipf(mflip[t]));
    exw[e] = ex;
    atomicAdd(&denom[t], ex);
}

// D lanes per edge, coalesced per-column scatter-add: agg[dst] += alpha * v[src]
template <int D>
__global__ __launch_bounds__(256) void edge_agg(
    const float* __restrict__ exw, const float* __restrict__ denom,
    const float* __restrict__ v, const int* __restrict__ src,
    const int* __restrict__ dst, float* __restrict__ agg, int E)
{
    constexpr int EPB = 256 / D;
    int le = threadIdx.x / D;
    int c  = threadIdx.x & (D - 1);
    int e  = blockIdx.x * EPB + le;
    if (e >= E) return;
    int s = src[e], t = dst[e];
    float w = exw[e] / denom[t];
    atomicAdd(&agg[(size_t)t * D + c], w * v[(size_t)s * D + c]);
}

// out = act(agg + skip); act 0 -> SiLU, 1 -> tanh
__global__ void finalize_k(
    const float* __restrict__ agg, const float* __restrict__ skip,
    float* __restrict__ out, long n, int act)
{
    long i = (long)blockIdx.x * blockDim.x + threadIdx.x;
    if (i >= n) return;
    float x = agg[i] + skip[i];
    out[i] = act ? tanhf(x) : (x / (1.0f + __expf(-x)));
}

// ---------------------------------------------------------------------------
// Host orchestration
// ---------------------------------------------------------------------------
extern "C" void kernel_launch(void* const* d_in, const int* in_sizes, int n_in,
                              void* d_out, int out_size, void* d_ws, size_t ws_size,
                              hipStream_t stream)
{
    // setup_inputs order: x, l1{Wq,bq,Wk,bk,Wv,bv,Ws,bs}, l2{...}, l3{...},
    // edge_src, edge_dst
    const int N = in_sizes[0] / DIN;
    const int E = in_sizes[25];
    const float* x    = (const float*)d_in[0];
    const int*   esrc = (const int*)d_in[25];
    const int*   edst = (const int*)d_in[26];

    float* f = (float*)d_ws;
    size_t nd = (size_t)N * 64;
    float* qb     = f; f += nd;
    float* kb     = f; f += nd;
    float* vb     = f; f += nd;
    float* sb     = f; f += nd;
    float* h      = f; f += nd;
    float* agg    = f; f += nd;
    float* logits = f; f += E;
    float* exw    = f; f += E;
    float* denom  = f; f += N;
    unsigned* mflip = (unsigned*)f; f += N;

    auto layer = [&](const float* hin, int pb, int dout, int act, float* hout) {
        const float* Wq = (const float*)d_in[pb+0]; const float* bq = (const float*)d_in[pb+1];
        const float* Wk = (const float*)d_in[pb+2]; const float* bk = (const float*)d_in[pb+3];
        const float* Wv = (const float*)d_in[pb+4]; const float* bv = (const float*)d_in[pb+5];
        const float* Wsk= (const float*)d_in[pb+6]; const float* bs = (const float*)d_in[pb+7];

        long na = (long)N * dout;
        if (dout == 64) {
            qkvs_gemm<64><<<(N + 31) / 32, 256, 0, stream>>>(
                hin, Wq, Wk, Wv, Wsk, bq, bk, bv, bs, qb, kb, vb, sb);
        } else {
            qkvs_gemm<32><<<(N + 63) / 64, 256, 0, stream>>>(
                hin, Wq, Wk, Wv, Wsk, bq, bk, bv, bs, qb, kb, vb, sb);
        }

        zero_f32<<<(int)((na + 255) / 256), 256, 0, stream>>>(agg, na);
        zero_f32<<<(N + 255) / 256, 256, 0, stream>>>(denom, N);
        zero_f32<<<(N + 255) / 256, 256, 0, stream>>>((float*)mflip, N); // 0 == -inf in flip space

        float scale = 1.0f / sqrtf((float)dout);  // d_head = dout (H=1)
        if (dout == 64) {
            edge_logits<64><<<(E + 7) / 8, 256, 0, stream>>>(
                qb, kb, esrc, edst, logits, mflip, E, scale);
        } else {
            edge_logits<32><<<(E + 7) / 8, 256, 0, stream>>>(
                qb, kb, esrc, edst, logits, mflip, E, scale);
        }
        edge_exp<<<(E + 255) / 256, 256, 0, stream>>>(
            logits, mflip, edst, exw, denom, E);
        if (dout == 64) {
            edge_agg<64><<<(E + 3) / 4, 256, 0, stream>>>(
                exw, denom, vb, esrc, edst, agg, E);
        } else {
            edge_agg<32><<<(E + 7) / 8, 256, 0, stream>>>(
                exw, denom, vb, esrc, edst, agg, E);
        }
        finalize_k<<<(int)((na + 255) / 256), 256, 0, stream>>>(
            agg, sb, hout, na, act);
    };

    layer(x, 1,  64, 0, h);                 // l1 + SiLU
    layer(h, 9,  64, 0, h);                 // l2 + SiLU (safe: GEMM reads h before finalize rewrites)
    layer(h, 17, 32, 1, (float*)d_out);     // l3 + tanh
}
